// Net_14224931684662
// MI455X (gfx1250) — matrix operand order repaired and checked by reading, not compiled, not run on hardware
//
#include <hip/hip_runtime.h>

#define NNODES   50000
#define NEDGES   800000
#define FDIM     256
#define NCLS     47
#define NCLS_PAD 64

typedef __attribute__((ext_vector_type(16))) __bf16 v16bf;
typedef __attribute__((ext_vector_type(8)))  __bf16 v8bf;
typedef __attribute__((ext_vector_type(8)))  float  v8f;

#define BM 128
#define BN 64
#define BK 32
#define APITCH 40   // bf16 elements -> 80B row pitch (20 banks; conflict-free b128 frag reads)
#define BPITCH 40

union FragU { v16bf v; v8bf h[2]; };

// ---------------- fp32 -> bf16 elementwise ----------------
__global__ __launch_bounds__(256)
void cvt_f32_bf16(const float* __restrict__ x, __bf16* __restrict__ y, int n) {
  int i = (blockIdx.x * 256 + threadIdx.x) * 4;
  if (i + 3 < n) {
    float4 f = *(const float4*)(x + i);
    y[i + 0] = (__bf16)f.x; y[i + 1] = (__bf16)f.y;
    y[i + 2] = (__bf16)f.z; y[i + 3] = (__bf16)f.w;
  } else {
    for (int q = i; q < n; ++q) y[q] = (__bf16)x[q];
  }
}

// ---------------- weight fp32[K][N] -> bf16 Wt[Npad][K] (N-major, zero-padded) ----------------
__global__ __launch_bounds__(256)
void transpose_w(const float* __restrict__ W, __bf16* __restrict__ Wt,
                 int K, int N, int Npad) {
  int id = blockIdx.x * 256 + threadIdx.x;
  if (id >= K * Npad) return;
  int n = id % Npad, k = id / Npad;
  Wt[n * K + k] = (n < N) ? (__bf16)W[k * N + n] : (__bf16)0.0f;
}

// ---------------- edge gather + segment-max scatter ----------------
// hp values are post-ReLU (>=0) so u32 bit-pattern max == float max; agg pre-zeroed
// matches reference's segment_max + neginf->0 semantics exactly.
__global__ __launch_bounds__(256)
void scatter_max_bf16(const __bf16* __restrict__ hp, const int* __restrict__ src,
                      const int* __restrict__ dst, unsigned int* __restrict__ agg,
                      int nE) {
  int gw   = (blockIdx.x * 256 + threadIdx.x) >> 5;   // one wave per edge
  int lane = threadIdx.x & 31;
  if (gw >= nE) return;
  int s = src[gw];
  int d = dst[gw];
  v8bf m = *(const v8bf*)(hp + (size_t)s * FDIM + lane * 8);   // global_load_b128
  unsigned int* ap = agg + (size_t)d * FDIM + lane * 8;
#pragma unroll
  for (int j = 0; j < 8; ++j) {
    float v = (float)m[j];
    atomicMax(ap + j, __float_as_uint(v));
  }
}

// ---------------- tiled WMMA GEMM: Out = [relu]( A0@W0 (+ A1@W1) + bias ) ----------------
// A0: bf16 [M][K] row-major; A1: fp32 [M][K] (converted to bf16 while staging);
// Wt*: bf16 [Ntiles*BN? >= n0+BN][K], N-major (pre-transposed).
template<bool HAS2, bool RELU, bool OUTBF16>
__global__ __launch_bounds__(256)
void sage_gemm(const __bf16* __restrict__ A0, const __bf16* __restrict__ Wt0,
               const float* __restrict__ A1, const __bf16* __restrict__ Wt1,
               const float* __restrict__ bias, void* __restrict__ Out,
               int M, int K, int Nout, int ldOut) {
  __shared__ __bf16 ldsA[BM * APITCH];
  __shared__ __bf16 ldsB[BN * BPITCH];

  const int tid  = threadIdx.x;
  const int lane = tid & 31;
  const int wave = tid >> 5;
  const int wm   = wave & 3;     // 4 waves across M: rows wm*32
  const int wn   = wave >> 2;    // 2 waves across N: cols wn*32
  const int m0   = blockIdx.x * BM;
  const int n0   = blockIdx.y * BN;

  v8f acc[2][2] = {};

  const int  laneM  = lane & 15;
  const bool hiHalf = lane >= 16;

  const int pairs = HAS2 ? 2 : 1;
  for (int p = 0; p < pairs; ++p) {
    const __bf16* Wt = (p == 0) ? Wt0 : Wt1;
    for (int kb = 0; kb < K; kb += BK) {
      // ---- stage A tile 128x32: each of 256 threads moves 16 bf16 ----
      {
        int row = tid >> 1;
        int kk  = (tid & 1) * 16;
        __bf16* dp = &ldsA[row * APITCH + kk];
        int gm = m0 + row;
        bool inb = gm < M;
        if (!HAS2 || p == 0) {
          if (inb) {
            const v8bf* sp = (const v8bf*)(A0 + (size_t)gm * K + kb + kk);
            ((v8bf*)dp)[0] = sp[0];
            ((v8bf*)dp)[1] = sp[1];
          } else {
            v8bf z = {};
            ((v8bf*)dp)[0] = z; ((v8bf*)dp)[1] = z;
          }
        } else {
          if (inb) {
            const float* sp = A1 + (size_t)gm * K + kb + kk;
#pragma unroll
            for (int q = 0; q < 16; ++q) dp[q] = (__bf16)sp[q];
          } else {
#pragma unroll
            for (int q = 0; q < 16; ++q) dp[q] = (__bf16)0.0f;
          }
        }
      }
      // ---- stage B tile 64x32 (N-major): threads 0..127, 16 bf16 each ----
      if (tid < 128) {
        int nrow = tid >> 1;
        int kk   = (tid & 1) * 16;
        const v8bf* sp = (const v8bf*)(Wt + (size_t)(n0 + nrow) * K + kb + kk);
        __bf16* dp = &ldsB[nrow * BPITCH + kk];
        ((v8bf*)dp)[0] = sp[0];
        ((v8bf*)dp)[1] = sp[1];
      }
      __syncthreads();

      // ---- assemble fragments (2x ds_load_b128 each) + 4 WMMAs ----
      FragU fa[2], fb[2];
#pragma unroll
      for (int i = 0; i < 2; ++i) {
        int row  = wm * 32 + i * 16 + laneM;
        int kOff = hiHalf ? 8 : 0;              // ISA 16-bit A layout
        const __bf16* base = &ldsA[row * APITCH + kOff];
        fa[i].h[0] = *(const v8bf*)(base);
        fa[i].h[1] = *(const v8bf*)(base + 16);
      }
#pragma unroll
      for (int j = 0; j < 2; ++j) {
        int nrow = wn * 32 + j * 16 + laneM;
        int k0   = hiHalf ? 8 : 0;              // B: same K order as A (ISA 16-bit layout)
        const __bf16* base = &ldsB[nrow * BPITCH + k0];
        fb[j].h[0] = *(const v8bf*)(base);
        fb[j].h[1] = *(const v8bf*)(base + 16);
      }
#pragma unroll
      for (int i = 0; i < 2; ++i)
#pragma unroll
        for (int j = 0; j < 2; ++j)
          acc[i][j] = __builtin_amdgcn_wmma_f32_16x16x32_bf16(
              false, fa[i].v, false, fb[j].v, (short)0, acc[i][j], false, false);
      __syncthreads();
    }
  }

  // ---- epilogue: C layout = lane%16 -> N col; VGPR r -> M row r (+8 for hi half) ----
  const int mAdd = hiHalf ? 8 : 0;
#pragma unroll
  for (int j = 0; j < 2; ++j) {
    int n = n0 + wn * 32 + j * 16 + laneM;
    float bv = (n < Nout) ? bias[n] : 0.0f;
#pragma unroll
    for (int i = 0; i < 2; ++i) {
#pragma unroll
      for (int r = 0; r < 8; ++r) {
        int gm = m0 + wm * 32 + i * 16 + r + mAdd;
        if (gm < M && n < Nout) {
          float v = acc[i][j][r] + bv;
          if (RELU) v = fmaxf(v, 0.0f);
          if (OUTBF16) ((__bf16*)Out)[(size_t)gm * ldOut + n] = (__bf16)v;
          else          ((float*)Out)[(size_t)gm * ldOut + n] = v;
        }
      }
    }
  }
}

extern "C" void kernel_launch(void* const* d_in, const int* in_sizes, int n_in,
                              void* d_out, int out_size, void* d_ws, size_t ws_size,
                              hipStream_t stream) {
  const float* features = (const float*)d_in[0];
  const int*   src      = (const int*)d_in[1];
  const int*   dst      = (const int*)d_in[2];
  const float* W_pool1  = (const float*)d_in[3];
  const float* b_pool1  = (const float*)d_in[4];
  const float* W_self1  = (const float*)d_in[5];
  const float* W_neigh1 = (const float*)d_in[6];
  const float* b1       = (const float*)d_in[7];
  const float* W_pool2  = (const float*)d_in[8];
  const float* b_pool2  = (const float*)d_in[9];
  const float* W_self2  = (const float*)d_in[10];
  const float* W_neigh2 = (const float*)d_in[11];
  const float* b2       = (const float*)d_in[12];

  char* ws = (char*)d_ws;
  size_t off = 0;
  auto take = [&](size_t bytes) -> char* {
    char* p = ws + off;
    off = (off + bytes + 255) & ~(size_t)255;
    return p;
  };
  __bf16* Xb  = (__bf16*)take((size_t)NNODES * FDIM * 2);
  __bf16* H   = (__bf16*)take((size_t)NNODES * FDIM * 2);
  __bf16* HP  = (__bf16*)take((size_t)NNODES * FDIM * 2);
  float*  AGG = (float*) take((size_t)NNODES * FDIM * 4);
  __bf16* Wp1t = (__bf16*)take((size_t)256 * 256 * 2);
  __bf16* Ws1t = (__bf16*)take((size_t)256 * 256 * 2);
  __bf16* Wn1t = (__bf16*)take((size_t)256 * 256 * 2);
  __bf16* Wp2t = (__bf16*)take((size_t)256 * 256 * 2);
  __bf16* Ws2t = (__bf16*)take((size_t)NCLS_PAD * 256 * 2);
  __bf16* Wn2t = (__bf16*)take((size_t)NCLS_PAD * 256 * 2);

  dim3 blk(256);

  // one-time-per-call conversions (cheap, deterministic)
  {
    int n = NNODES * FDIM;
    cvt_f32_bf16<<<(n / 4 + 255) / 256, blk, 0, stream>>>(features, Xb, n);
  }
  transpose_w<<<(256 * 256 + 255) / 256, blk, 0, stream>>>(W_pool1,  Wp1t, 256, 256, 256);
  transpose_w<<<(256 * 256 + 255) / 256, blk, 0, stream>>>(W_self1,  Ws1t, 256, 256, 256);
  transpose_w<<<(256 * 256 + 255) / 256, blk, 0, stream>>>(W_neigh1, Wn1t, 256, 256, 256);
  transpose_w<<<(256 * 256 + 255) / 256, blk, 0, stream>>>(W_pool2,  Wp2t, 256, 256, 256);
  transpose_w<<<(256 * NCLS_PAD + 255) / 256, blk, 0, stream>>>(W_self2,  Ws2t, 256, NCLS, NCLS_PAD);
  transpose_w<<<(256 * NCLS_PAD + 255) / 256, blk, 0, stream>>>(W_neigh2, Wn2t, 256, NCLS, NCLS_PAD);

  dim3 gFull((NNODES + BM - 1) / BM, FDIM / BN);  // 391 x 4
  dim3 gOut ((NNODES + BM - 1) / BM, 1);          // 391 x 1 (N=47 padded to 64)
  const int scatterBlocks = (NEDGES * 32) / 256;  // one wave per edge

  // ---- layer 1 ----
  sage_gemm<false, true, true><<<gFull, blk, 0, stream>>>(
      Xb, Wp1t, nullptr, nullptr, b_pool1, HP, NNODES, FDIM, FDIM, FDIM);
  hipMemsetAsync(AGG, 0, (size_t)NNODES * FDIM * 4, stream);
  scatter_max_bf16<<<scatterBlocks, blk, 0, stream>>>(HP, src, dst, (unsigned int*)AGG, NEDGES);
  sage_gemm<true, true, true><<<gFull, blk, 0, stream>>>(
      Xb, Ws1t, AGG, Wn1t, b1, H, NNODES, FDIM, FDIM, FDIM);

  // ---- layer 2 ----
  sage_gemm<false, true, true><<<gFull, blk, 0, stream>>>(
      H, Wp2t, nullptr, nullptr, b_pool2, HP, NNODES, FDIM, FDIM, FDIM);
  hipMemsetAsync(AGG, 0, (size_t)NNODES * FDIM * 4, stream);
  scatter_max_bf16<<<scatterBlocks, blk, 0, stream>>>(HP, src, dst, (unsigned int*)AGG, NEDGES);
  sage_gemm<true, false, false><<<gOut, blk, 0, stream>>>(
      H, Ws2t, AGG, Wn2t, b2, d_out, NNODES, FDIM, NCLS, NCLS);

  (void)in_sizes; (void)n_in; (void)out_size; (void)ws_size;
}
